// APNB_33947421508408
// MI455X (gfx1250) — matrix operand order repaired and checked by reading, not compiled, not run on hardware
//
#include <hip/hip_runtime.h>
#include <hip/hip_bf16.h>

typedef __bf16 bf16;
typedef __attribute__((ext_vector_type(16))) __bf16 v16bf;
typedef __attribute__((ext_vector_type(8)))  __bf16 v8bf;
typedef __attribute__((ext_vector_type(8)))  float  v8f;

#define BM 128
#define BN 128
#define BK 32
#define LS 40  // LDS row stride in halves (BK + 8 pad): 80B rows, 16B aligned

enum { EPI_NONE = 0, EPI_BIAS = 1, EPI_BNRELU = 2 };

__device__ __forceinline__ v8f wmma_bf16(v16bf a, v16bf b, v8f c) {
  return __builtin_amdgcn_wmma_f32_16x16x32_bf16(
      /*neg_a=*/false, a, /*neg_b=*/false, b,
      /*c_mod=*/(short)0, c, /*reuse_a=*/false, /*reuse_b=*/false);
}

// A fragment: lane m fixed, elements 0..7 -> k=kaoff+e, 8..15 -> k=kaoff+16+e.
// With [m][k] K-contiguous LDS rows this is two ds_load_b128.
__device__ __forceinline__ v16bf frag_a(const bf16* row, int kaoff) {
  v8bf lo = *(const v8bf*)(row + kaoff);
  v8bf hi = *(const v8bf*)(row + 16 + kaoff);
  return __builtin_shufflevector(lo, hi, 0, 1, 2, 3, 4, 5, 6, 7,
                                 8, 9, 10, 11, 12, 13, 14, 15);
}
// B fragment: lane n fixed, elements 0..7 -> k=kboff+e, 8..15 -> k=kboff+16+e.
__device__ __forceinline__ v16bf frag_b(const bf16* row, int kboff) {
  v8bf lo = *(const v8bf*)(row + kboff);
  v8bf hi = *(const v8bf*)(row + 16 + kboff);
  return __builtin_shufflevector(lo, hi, 0, 1, 2, 3, 4, 5, 6, 7,
                                 8, 9, 10, 11, 12, 13, 14, 15);
}

__device__ __forceinline__ float rmax16(float v) {
  #pragma unroll
  for (int m = 1; m < 16; m <<= 1) v = fmaxf(v, __shfl_xor(v, m, 32));
  return v;
}
__device__ __forceinline__ float rsum16(float v) {
  #pragma unroll
  for (int m = 1; m < 16; m <<= 1) v += __shfl_xor(v, m, 32);
  return v;
}

// ---------------------------------------------------------------------------
// TN WMMA GEMM: D[m,n] = epi( sum_k A[m*lda+k] * B[n*ldb+k] )
//   A: row-major M x K (K-contiguous).  B: row-major N x K (pixel-major).
//   BSPLIT: B rows use B0 for k<ksplit else B1 (virtual concat along K).
//   DT (bf16 out): store pixel-major D[n*ldd+m], packed 8-bf16 b128 stores.
//   !DT (f32 out): store row-major D[m*ldd+n].
// 256 threads = 8 waves; block tile 128x128; wave tile 32x64; K-step 32.
// ---------------------------------------------------------------------------
template <typename OutT, int EPI, bool BSPLIT, bool DT>
__global__ __launch_bounds__(256) void gemm_wmma(
    const bf16* __restrict__ A, long sA, int lda,
    const bf16* __restrict__ B0, const bf16* __restrict__ B1, long sB, int ldb,
    int ksplit,
    OutT* __restrict__ D, long sD, int ldd,
    int M, int N, int K,
    const float* __restrict__ bias, const float* __restrict__ gamma,
    const float* __restrict__ beta, const float* __restrict__ mean,
    const float* __restrict__ var) {
  __shared__ bf16 lA[BM * LS];
  __shared__ bf16 lB[BN * LS];

  const int b  = blockIdx.z;
  const int m0 = blockIdx.x * BM;
  const int n0 = blockIdx.y * BN;
  const bf16* Ab  = A + (long)b * sA;
  const bf16* B0b = B0 + (long)b * sB;
  const bf16* B1b = BSPLIT ? (B1 + (long)b * sB) : nullptr;

  const int tid  = threadIdx.x;
  const int lane = tid & 31;
  const int wave = tid >> 5;
  const int wm   = (wave & 3) * 32;   // 4 waves along M
  const int wn   = (wave >> 2) * 64;  // 2 waves along N
  const int ml    = lane & 15;
  const int kaoff = (lane < 16) ? 0 : 8;
  const int kboff = (lane < 16) ? 0 : 8;
  const int mhalf = (lane < 16) ? 0 : 8;

  v8f acc[2][4];
  const v8f vzero = {0.f, 0.f, 0.f, 0.f, 0.f, 0.f, 0.f, 0.f};
  #pragma unroll
  for (int i = 0; i < 2; ++i)
    #pragma unroll
    for (int j = 0; j < 4; ++j) acc[i][j] = vzero;

  for (int k0 = 0; k0 < K; k0 += BK) {
    __syncthreads();
    // stage A tile: 128 rows x 4 chunks of 8 bf16 -> [m][k] (b128 in/out)
    #pragma unroll
    for (int i = 0; i < 2; ++i) {
      int c = tid * 2 + i, m = c >> 2, kq = c & 3;
      *(uint4*)(&lA[m * LS + kq * 8]) =
          *(const uint4*)(Ab + (long)(m0 + m) * lda + k0 + kq * 8);
    }
    // stage B tile: 128 rows x 4 chunks -> [n][k]
    #pragma unroll
    for (int i = 0; i < 2; ++i) {
      int c = tid * 2 + i, n = c >> 2, kq = c & 3;
      int kg = k0 + kq * 8;
      const bf16* bp = (!BSPLIT || kg < ksplit)
                           ? (B0b + (long)(n0 + n) * ldb + kg)
                           : (B1b + (long)(n0 + n) * ldb + (kg - ksplit));
      *(uint4*)(&lB[n * LS + kq * 8]) = *(const uint4*)bp;
    }
    if (k0 + BK < K)  // next-tile prefetch (global_prefetch_b8)
      __builtin_prefetch(Ab + (long)(m0 + tid / 2) * lda + k0 + BK, 0, 1);
    __syncthreads();

    v16bf af[2], bfv[4];
    #pragma unroll
    for (int i = 0; i < 2; ++i)
      af[i] = frag_a(&lA[(wm + i * 16 + ml) * LS], kaoff);
    #pragma unroll
    for (int j = 0; j < 4; ++j)
      bfv[j] = frag_b(&lB[(wn + j * 16 + ml) * LS], kboff);
    #pragma unroll
    for (int i = 0; i < 2; ++i)
      #pragma unroll
      for (int j = 0; j < 4; ++j) acc[i][j] = wmma_bf16(af[i], bfv[j], acc[i][j]);
  }

  // ---- epilogue ----
  OutT* Db = D + (long)b * sD;
  #pragma unroll
  for (int i = 0; i < 2; ++i) {
    const int mb = m0 + wm + i * 16 + mhalf;
    #pragma unroll
    for (int j = 0; j < 4; ++j) {
      const int n = n0 + wn + j * 16 + ml;
      if constexpr (DT) {
        v8bf o;
        #pragma unroll
        for (int r = 0; r < 8; ++r) {
          int m = mb + r;
          float v = acc[i][j][r];
          if constexpr (EPI == EPI_BIAS) v += bias[m];
          if constexpr (EPI == EPI_BNRELU) {
            float inv = rsqrtf(var[m] + 1e-5f);
            float s = gamma[m] * inv;
            float t = (bias[m] - mean[m]) * s + beta[m];
            v = fmaxf(v * s + t, 0.f);
          }
          o[r] = (bf16)v;
        }
        *(v8bf*)((bf16*)Db + (long)n * ldd + mb) = o;  // pixel-major b128 store
      } else {
        #pragma unroll
        for (int r = 0; r < 8; ++r) {
          int m = mb + r;
          float v = acc[i][j][r];
          if constexpr (EPI == EPI_BIAS) v += bias[m];
          if constexpr (EPI == EPI_BNRELU) {
            float inv = rsqrtf(var[m] + 1e-5f);
            float s = gamma[m] * inv;
            float t = (bias[m] - mean[m]) * s + beta[m];
            v = fmaxf(v * s + t, 0.f);
          }
          Db[(long)m * ldd + n] = (OutT)v;
        }
      }
    }
  }
}

// ---------------------------------------------------------------------------
// Fused sim + softmax. M=128 pixels/block (8 waves x 16), N=S(112), K=256.
// A = kf pixel-major (9216 x 256), B = keypT (112 x 256) both K-contiguous.
// Softmax over 110 valid cols; attn (9216 x 128) bf16, pads zeroed.
// ---------------------------------------------------------------------------
__global__ __launch_bounds__(256) void sim_softmax_kernel(
    const bf16* __restrict__ kf, long skf,
    const bf16* __restrict__ keypT, long skeyp,
    bf16* __restrict__ attn, long sattn) {
  __shared__ bf16 lA[128 * LS];
  __shared__ bf16 lB[112 * LS];

  const int b  = blockIdx.y;
  const int p0 = blockIdx.x * 128;
  const int tid = threadIdx.x, lane = tid & 31, wave = tid >> 5;
  const int ml = lane & 15;
  const int kaoff = (lane < 16) ? 0 : 8;
  const int kboff = (lane < 16) ? 0 : 8;
  const int mhalf = (lane < 16) ? 0 : 8;

  const bf16* kfb = kf + (long)b * skf;
  const bf16* kpb = keypT + (long)b * skeyp;

  v8f acc[7];
  const v8f vzero = {0.f, 0.f, 0.f, 0.f, 0.f, 0.f, 0.f, 0.f};
  #pragma unroll
  for (int t = 0; t < 7; ++t) acc[t] = vzero;

  for (int k0 = 0; k0 < 256; k0 += BK) {
    __syncthreads();
    #pragma unroll
    for (int i = 0; i < 2; ++i) {  // A: 128 x 32, K-contiguous chunks
      int c = tid * 2 + i, m = c >> 2, kq = c & 3;
      *(uint4*)(&lA[m * LS + kq * 8]) =
          *(const uint4*)(kfb + (long)(p0 + m) * 256 + k0 + kq * 8);
    }
    #pragma unroll
    for (int i = 0; i < 2; ++i) {  // B: 112 x 32 (448 chunks)
      int c = tid * 2 + i;
      if (c < 448) {
        int n = c >> 2, kq = c & 3;
        *(uint4*)(&lB[n * LS + kq * 8]) =
            *(const uint4*)(kpb + (long)n * 256 + k0 + kq * 8);
      }
    }
    __syncthreads();

    v16bf af = frag_a(&lA[(wave * 16 + ml) * LS], kaoff);
    #pragma unroll
    for (int t = 0; t < 7; ++t) {
      v16bf bfv = frag_b(&lB[(t * 16 + ml) * LS], kboff);
      acc[t] = wmma_bf16(af, bfv, acc[t]);
    }
  }

  // softmax per pixel row (both lane-halves carry independent rows)
  const float scale = 0.0625f;  // 1/sqrt(256)
  #pragma unroll
  for (int r = 0; r < 8; ++r) {
    float v[7];
    float mx = -3.0e38f;
    #pragma unroll
    for (int t = 0; t < 7; ++t) {
      int n = t * 16 + ml;
      v[t] = (n < 110) ? acc[t][r] * scale : -3.0e38f;
      mx = fmaxf(mx, v[t]);
    }
    mx = rmax16(mx);
    float sum = 0.f;
    #pragma unroll
    for (int t = 0; t < 7; ++t) {
      v[t] = (t * 16 + ml < 110) ? __expf(v[t] - mx) : 0.f;
      sum += v[t];
    }
    sum = rsum16(sum);
    float rinv = 1.f / sum;
    int p = p0 + wave * 16 + r + mhalf;
    bf16* row = attn + (long)b * sattn + (long)p * 128;
    #pragma unroll
    for (int t = 0; t < 7; ++t) row[t * 16 + ml] = (bf16)(v[t] * rinv);
    row[112 + ml] = (bf16)0.f;  // pad cols 112..127
  }
}

// ---------------------------------------------------------------------------
// Pyramid pooling over pixel-major input X[p*CH + c]. One block per (c,b).
// TR=true: write P[s*ldo + c] (keypT), else P[c*ldo + s] (poolV). Pads zeroed.
// ---------------------------------------------------------------------------
template <typename OutT, bool TR>
__global__ void psp_pool_kernel(const bf16* __restrict__ X, long sX, int CH,
                                OutT* __restrict__ P, long sP, int ldo) {
  const int c = blockIdx.x, b = blockIdx.y, tid = threadIdx.x;
  if (tid >= 112) return;
  OutT* out = P + (long)b * sP + (TR ? ((long)tid * ldo + c) : ((long)c * ldo + tid));
  if (tid >= 110) { *out = (OutT)0.f; return; }
  int s, cell;
  if (tid == 0)       { s = 1; cell = 0; }
  else if (tid < 10)  { s = 3; cell = tid - 1; }
  else if (tid < 46)  { s = 6; cell = tid - 10; }
  else                { s = 8; cell = tid - 46; }
  const int gy = cell / s, gx = cell % s, hh = 96 / s;
  const bf16* x = X + (long)b * sX + c;
  float sum = 0.f;
  for (int y = 0; y < hh; ++y)
    for (int xx = 0; xx < hh; ++xx)
      sum += (float)x[(long)((gy * hh + y) * 96 + gx * hh + xx) * CH];
  *out = (OutT)(sum / (float)(hh * hh));
}

// valueT[c,s] = Wv[c,:]·poolV[:,s] + bv[c]  (pool-then-project; tiny GEMM)
__global__ __launch_bounds__(256) void value_kernel(
    const float* __restrict__ poolV, long sPV, const float* __restrict__ Wv,
    const float* __restrict__ bv, bf16* __restrict__ valueT, long sV) {
  const int s = blockIdx.x, b = blockIdx.y, c = threadIdx.x;
  float v = 0.f;
  if (s < 110) {
    v = bv[c];
    const float* pv = poolV + (long)b * sPV;
    for (int ci = 0; ci < 512; ++ci) v += Wv[c * 512 + ci] * pv[ci * 112 + s];
  }
  valueT[(long)b * sV + (long)c * 128 + s] = (bf16)v;
}

__global__ void cvt_f32_bf16(const float* __restrict__ src,
                             bf16* __restrict__ dst, long n) {
  long i = (long)blockIdx.x * blockDim.x + threadIdx.x;
  long stride = (long)gridDim.x * blockDim.x;
  for (; i < n; i += stride) dst[i] = (bf16)src[i];
}

// Transpose + convert: src [C][NP] f32 (channel-major) -> dst [NP][C] bf16
// (pixel-major). 64x64 tiles through LDS; coalesced global loads and stores.
__global__ __launch_bounds__(256) void transpose_cvt_kernel(
    const float* __restrict__ src, bf16* __restrict__ dst, int C, int NP) {
  __shared__ bf16 lt[64][80];  // [p][c], padded rows (160B, 16B aligned)
  const int b = blockIdx.z, c0 = blockIdx.y * 64, p0 = blockIdx.x * 64;
  const float* sb = src + (long)b * C * NP;
  bf16* db = dst + (long)b * NP * C;
  const int tid = threadIdx.x;
  #pragma unroll
  for (int i = 0; i < 4; ++i) {  // 64 c-rows x 16 float4 chunks
    int ch = i * 256 + tid, cc = ch >> 4, pq = ch & 15;
    float4 v = *(const float4*)(sb + (long)(c0 + cc) * NP + p0 + pq * 4);
    lt[pq * 4 + 0][cc] = (bf16)v.x;
    lt[pq * 4 + 1][cc] = (bf16)v.y;
    lt[pq * 4 + 2][cc] = (bf16)v.z;
    lt[pq * 4 + 3][cc] = (bf16)v.w;
  }
  __syncthreads();
  #pragma unroll
  for (int i = 0; i < 2; ++i) {  // 64 p-rows x 8 chunks of 8 bf16
    int ch = i * 256 + tid, pp = ch >> 3, cq = ch & 7;
    *(v8bf*)(db + (long)(p0 + pp) * C + c0 + cq * 8) = *(v8bf*)&lt[pp][cq * 8];
  }
}

// ---------------------------------------------------------------------------
extern "C" void kernel_launch(void* const* d_in, const int* in_sizes, int n_in,
                              void* d_out, int out_size, void* d_ws,
                              size_t ws_size, hipStream_t stream) {
  (void)in_sizes; (void)n_in; (void)out_size; (void)ws_size;
  const float* feats = (const float*)d_in[0];
  const float* Wk = (const float*)d_in[1];
  const float* bk = (const float*)d_in[2];
  const float* gk = (const float*)d_in[3];
  const float* betak = (const float*)d_in[4];
  const float* mk = (const float*)d_in[5];
  const float* vk = (const float*)d_in[6];
  const float* Wv = (const float*)d_in[7];
  const float* bv = (const float*)d_in[8];
  const float* Ww = (const float*)d_in[9];
  const float* bw = (const float*)d_in[10];
  const float* Wo = (const float*)d_in[11];
  const float* bo = (const float*)d_in[12];
  const float* go = (const float*)d_in[13];
  const float* betao = (const float*)d_in[14];
  const float* mo = (const float*)d_in[15];
  const float* vo = (const float*)d_in[16];
  float* out = (float*)d_out;

  const int B = 8, NP = 9216, CIN = 512, CK = 256, CV = 256, COUT = 512;
  const int SP = 112, SP2 = 128;

  char* ws = (char*)d_ws;
  size_t off = 0;
  auto take = [&](size_t bytes) -> void* {
    void* p = ws + off;
    off = (off + bytes + 255) & ~(size_t)255;
    return p;
  };
  // all activations pixel-major [pixel][channel]
  bf16*  featsB = (bf16*) take((size_t)B * NP * CIN * 2);
  bf16*  kf     = (bf16*) take((size_t)B * NP * CK * 2);
  bf16*  attn   = (bf16*) take((size_t)B * NP * SP2 * 2);
  bf16*  ctx    = (bf16*) take((size_t)B * NP * CV * 2);
  bf16*  ctxw   = (bf16*) take((size_t)B * NP * COUT * 2);
  bf16*  keypT  = (bf16*) take((size_t)B * SP * CK * 2);   // [s][c]
  float* poolV  = (float*)take((size_t)B * CIN * SP * 4);  // [c][s]
  bf16*  valueT = (bf16*) take((size_t)B * CV * SP2 * 2);  // [c][s]
  bf16*  WkB    = (bf16*) take((size_t)CK * CIN * 2);
  bf16*  WwB    = (bf16*) take((size_t)COUT * CV * 2);
  bf16*  WoB    = (bf16*) take((size_t)COUT * 2 * CIN * 2);

  // 1) staging: feats -> pixel-major bf16 (tiled transpose); weights -> bf16
  transpose_cvt_kernel<<<dim3(NP / 64, CIN / 64, B), 256, 0, stream>>>(
      feats, featsB, CIN, NP);
  cvt_f32_bf16<<<512, 256, 0, stream>>>(Wk, WkB, (long)CK * CIN);
  cvt_f32_bf16<<<512, 256, 0, stream>>>(Ww, WwB, (long)COUT * CV);
  cvt_f32_bf16<<<2048, 256, 0, stream>>>(Wo, WoB, (long)COUT * 2 * CIN);

  // 2) kf = ReLU(BN(Wk @ feats))  -> pixel-major [9216 x 256]
  gemm_wmma<bf16, EPI_BNRELU, false, true>
      <<<dim3(CK / BM, NP / BN, B), 256, 0, stream>>>(
          WkB, 0, CIN, featsB, nullptr, (long)NP * CIN, CIN, 0,
          kf, (long)NP * CK, CK, CK, NP, CIN, bk, gk, betak, mk, vk);

  // 3) pyramid pools: keypT (bf16 [s][c]) from kf, poolV (f32 [c][s]) from feats
  psp_pool_kernel<bf16, true><<<dim3(CK, B), 128, 0, stream>>>(
      kf, (long)NP * CK, CK, keypT, (long)SP * CK, CK);
  psp_pool_kernel<float, false><<<dim3(CIN, B), 128, 0, stream>>>(
      featsB, (long)NP * CIN, CIN, poolV, (long)CIN * SP, SP);

  // 4) valueT = (Wv @ psp(feats) + bv)  [256 x 128, s-pads zeroed]
  value_kernel<<<dim3(SP2, B), 256, 0, stream>>>(
      poolV, (long)CIN * SP, Wv, bv, valueT, (long)CV * SP2);

  // 5) fused sim + softmax -> attn [9216 x 128] pixel-major
  sim_softmax_kernel<<<dim3(NP / 128, B), 256, 0, stream>>>(
      kf, (long)NP * CK, keypT, (long)SP * CK, attn, (long)NP * SP2);

  // 6) ctx = value^T @ attn^T : M=256(ch), N=9216(pix), K=128 -> pixel-major
  gemm_wmma<bf16, EPI_NONE, false, true>
      <<<dim3(CV / BM, NP / BN, B), 256, 0, stream>>>(
          valueT, (long)CV * SP2, SP2, attn, nullptr, (long)NP * SP2, SP2, 0,
          ctx, (long)NP * CV, CV, CV, NP, SP2,
          nullptr, nullptr, nullptr, nullptr, nullptr);

  // 7) ctxw = Ww @ ctx + bw : M=512, N=9216, K=256 -> pixel-major
  gemm_wmma<bf16, EPI_BIAS, false, true>
      <<<dim3(COUT / BM, NP / BN, B), 256, 0, stream>>>(
          WwB, 0, CV, ctx, nullptr, (long)NP * CV, CV, 0,
          ctxw, (long)NP * COUT, COUT, COUT, NP, CV,
          bw, nullptr, nullptr, nullptr, nullptr);

  // 8) out = ReLU(BN(Wo @ [ctxw; feats])) : K=1024 split, f32 channel-major
  gemm_wmma<float, EPI_BNRELU, true, false>
      <<<dim3(COUT / BM, NP / BN, B), 256, 0, stream>>>(
          WoB, 0, 2 * CIN, ctxw, featsB, (long)NP * COUT, CIN, COUT,
          out, (long)COUT * NP, NP, COUT, NP, 2 * CIN, bo, go, betao, mo, vo);
}
